// Net_43860206026847
// MI455X (gfx1250) — matrix operand order repaired and checked by reading, not compiled, not run on hardware
//
#include <hip/hip_runtime.h>

#define B_ 64
#define S_ 512
#define D_ 768
#define H_ 1024

#if defined(__has_builtin)
#  if __has_builtin(__builtin_amdgcn_global_load_async_to_lds_b128) && \
      __has_builtin(__builtin_amdgcn_s_wait_asynccnt)
#    define USE_ASYNC_LDS 1
#  else
#    define USE_ASYNC_LDS 0
#  endif
#else
#  define USE_ASYNC_LDS 0
#endif

typedef __attribute__((ext_vector_type(16))) __bf16 v16bf;
typedef __attribute__((ext_vector_type(8)))  float  v8f;
typedef __attribute__((ext_vector_type(8)))  unsigned short u16x8;
typedef __attribute__((ext_vector_type(4)))  int    v4i;

#if USE_ASYNC_LDS
typedef __attribute__((address_space(1))) v4i* gv4i_p;   // global int4*
typedef __attribute__((address_space(3))) v4i* lv4i_p;   // LDS int4*
#endif

union BF16x16 {
    v16bf  v;
    u16x8  h[2];
    unsigned short u[16];
};

__device__ __forceinline__ unsigned short f2bf(float f) {
    __bf16 b = (__bf16)f;                    // native v_cvt (RTNE)
    return __builtin_bit_cast(unsigned short, b);
}

__device__ __forceinline__ v8f wmma_bf16(v16bf a, v16bf b, v8f c) {
    return __builtin_amdgcn_wmma_f32_16x16x32_bf16(
        false, a, false, b, (short)0, c, false, false);
}

// ---- fragment loaders -------------------------------------------------------
// A (16x32 bf16, MxK): lane r=lane&15 -> row M=r; g=lane>>4 -> K={8g..8g+7, 16+8g..+7}
__device__ __forceinline__ v16bf load_a_bf16(const unsigned short* A, int ld,
                                             int m0, int k0, int lane) {
    int r = lane & 15, g = lane >> 4;
    const unsigned short* p = A + (size_t)(m0 + r) * ld + k0 + (g << 3);
    BF16x16 a;
    a.h[0] = *(const u16x8*)p;
    a.h[1] = *(const u16x8*)(p + 16);
    return a.v;
}

__device__ __forceinline__ v16bf load_a_lds(const unsigned short* Alds, int k0, int lane) {
    int r = lane & 15, g = lane >> 4;
    const unsigned short* p = Alds + r * H_ + k0 + (g << 3);
    BF16x16 a;
    a.h[0] = *(const u16x8*)p;
    a.h[1] = *(const u16x8*)(p + 16);
    return a.v;
}

__device__ __forceinline__ v16bf load_a_f32(const float* A, size_t rowstride,
                                            int m0, int k0, int lane) {
    int r = lane & 15, g = lane >> 4;
    const float* p = A + (size_t)(m0 + r) * rowstride + k0 + (g << 3);
    BF16x16 a;
#pragma unroll
    for (int i = 0; i < 8; i++) a.u[i]     = f2bf(p[i]);
#pragma unroll
    for (int i = 0; i < 8; i++) a.u[8 + i] = f2bf(p[16 + i]);
    return a.v;
}

// B (32x16 bf16, KxN), pre-packed: frag[tile*512 + lane*16 + j] = W[n0+(lane&15)][k0+8*(lane>>4)+(j&7)+16*(j>>3)]
__device__ __forceinline__ v16bf load_b_frag(const unsigned short* F, int tile, int lane) {
    const unsigned short* p = F + (size_t)tile * 512 + lane * 16;
    BF16x16 b;
    b.h[0] = *(const u16x8*)p;
    b.h[1] = *(const u16x8*)(p + 8);
    return b.v;
}

// ---- stage a contiguous 32KB (16 x H bf16) tile into LDS -------------------
__device__ __forceinline__ void stage_tile_32k(unsigned short* lds,
                                               const unsigned short* gsrc, int tid) {
    u16x8* ld8 = (u16x8*)lds;
    const u16x8* g8 = (const u16x8*)gsrc;
#pragma unroll
    for (int i = 0; i < 8; i++) {
        int c = tid + (i << 8);
#if USE_ASYNC_LDS
        v4i* gp = (v4i*)(void*)(g8 + c);     // strip const, retype (generic AS)
        v4i* lp = (v4i*)(void*)(ld8 + c);
        __builtin_amdgcn_global_load_async_to_lds_b128(
            (gv4i_p)gp, (lv4i_p)lp, 0, 0);
#else
        ld8[c] = g8[c];
#endif
    }
#if USE_ASYNC_LDS
    __builtin_amdgcn_s_wait_asynccnt(0);
#endif
}

// ---- weight f32 -> bf16 fragment packing -----------------------------------
__global__ void wfrag_kernel(const float* __restrict__ W, int ldw, int coloff,
                             int N, int K, unsigned short* __restrict__ dst) {
    int idx = blockIdx.x * blockDim.x + threadIdx.x;
    if (idx >= N * K) return;
    int tile = idx >> 9, rem = idx & 511;
    int lane = rem >> 4, j = rem & 15;
    int ktiles = K >> 5;
    int n = ((tile / ktiles) << 4) + (lane & 15);
    int k = ((tile % ktiles) << 5) + ((lane >> 4) << 3) + (j & 7) + ((j >> 3) << 4);
    dst[idx] = f2bf(W[(size_t)n * ldw + coloff + k]);
}

// ---- h0 = x[:,0,:] @ Wh.T + bh ---------------------------------------------
__global__ void h0_kernel(const float* __restrict__ x,
                          const unsigned short* __restrict__ WhF,
                          const float* __restrict__ bh,
                          float* __restrict__ hF32,
                          unsigned short* __restrict__ hBF) {
    int lane = threadIdx.x & 31, w = threadIdx.x >> 5;
    int wgid = blockIdx.x * 8 + w;
    int m0 = (wgid >> 4) << 4;
    int n0 = (wgid & 15) << 6;
    v8f acc[4] = {};
    const int ktiles = D_ >> 5;
    v16bf bc[4], bn[4];
#pragma unroll
    for (int c = 0; c < 4; c++) bc[c] = load_b_frag(WhF, ((n0 >> 4) + c) * ktiles, lane);
    for (int kt = 0; kt < ktiles; kt++) {
        int ktn = (kt + 1 < ktiles) ? kt + 1 : kt;
#pragma unroll
        for (int c = 0; c < 4; c++) bn[c] = load_b_frag(WhF, ((n0 >> 4) + c) * ktiles + ktn, lane);
        v16bf a = load_a_f32(x, (size_t)S_ * D_, m0, kt << 5, lane);
#pragma unroll
        for (int c = 0; c < 4; c++) acc[c] = wmma_bf16(a, bc[c], acc[c]);
#pragma unroll
        for (int c = 0; c < 4; c++) bc[c] = bn[c];
    }
    int nl = lane & 15, gm = (lane >> 4) << 3;
#pragma unroll
    for (int c = 0; c < 4; c++) {
        int n = n0 + (c << 4) + nl;
        float bias = bh[n];
#pragma unroll
        for (int v = 0; v < 8; v++) {
            int m = m0 + gm + v;
            float val = acc[c][v] + bias;
            hF32[m * H_ + n] = val;
            hBF[m * H_ + n] = f2bf(val);
        }
    }
}

// ---- u[s,b,:] = x[b,s,:] @ Wm.T + bm  (m = s*64+b) -------------------------
__global__ void ugemm_kernel(const float* __restrict__ x,
                             const unsigned short* __restrict__ WmF,
                             const float* __restrict__ bm,
                             unsigned short* __restrict__ u) {
    __shared__ __align__(16) unsigned short ldsA[32 * 32];
    int tid = threadIdx.x, lane = tid & 31, w = tid >> 5;
    int m_base = blockIdx.y << 5;
    int n_base = blockIdx.x << 8;
    int wm = w >> 2, wn = w & 3;
    int m0 = m_base + (wm << 4);
    int n0 = n_base + (wn << 6);
    v8f acc[4] = {};
    const int ktiles = D_ >> 5;

    int e = tid << 2;
    int srow = e >> 5, scol = e & 31;
    int gmr = m_base + srow;
    int sb = gmr & 63, ss = gmr >> 6;
    const float* xrow = x + ((size_t)sb * S_ + ss) * D_;

    v16bf bc[4], bn[4];
#pragma unroll
    for (int c = 0; c < 4; c++) bc[c] = load_b_frag(WmF, ((n0 >> 4) + c) * ktiles, lane);

    for (int kt = 0; kt < ktiles; kt++) {
        int ktn = (kt + 1 < ktiles) ? kt + 1 : kt;
#pragma unroll
        for (int c = 0; c < 4; c++) bn[c] = load_b_frag(WmF, ((n0 >> 4) + c) * ktiles + ktn, lane);

        __syncthreads();
        float4 xv = *(const float4*)(xrow + (kt << 5) + scol);
        unsigned short* d = &ldsA[srow * 32 + scol];
        d[0] = f2bf(xv.x); d[1] = f2bf(xv.y); d[2] = f2bf(xv.z); d[3] = f2bf(xv.w);
        __syncthreads();

        int r = lane & 15, g = lane >> 4;
        const unsigned short* p = &ldsA[((wm << 4) + r) * 32 + (g << 3)];
        BF16x16 af;
        af.h[0] = *(const u16x8*)p;
        af.h[1] = *(const u16x8*)(p + 16);
#pragma unroll
        for (int c = 0; c < 4; c++) acc[c] = wmma_bf16(af.v, bc[c], acc[c]);
#pragma unroll
        for (int c = 0; c < 4; c++) bc[c] = bn[c];
    }
    int nl = lane & 15, gm = (lane >> 4) << 3;
#pragma unroll
    for (int c = 0; c < 4; c++) {
        int n = n0 + (c << 4) + nl;
        float bias = bm[n];
#pragma unroll
        for (int v = 0; v < 8; v++) {
            int m = m0 + gm + v;
            int s = m >> 6, b = m & 63;
            u[((size_t)s * B_ + b) * H_ + n] = f2bf(acc[c][v] + bias);
        }
    }
}

// ---- step 1: z & r gates; wave = 16(M) x 16(N), both gates ------------------
__global__ void step1_kernel(const unsigned short* __restrict__ u_t,
                             const unsigned short* __restrict__ hBF,
                             const float* __restrict__ hF32,
                             const unsigned short* __restrict__ WzU,
                             const unsigned short* __restrict__ WzH,
                             const unsigned short* __restrict__ WrU,
                             const unsigned short* __restrict__ WrH,
                             const float* __restrict__ bz,
                             const float* __restrict__ br,
                             float* __restrict__ zbuf,
                             unsigned short* __restrict__ rh) {
    __shared__ __align__(16) unsigned short ldsU[16 * H_];
    int tid = threadIdx.x, lane = tid & 31, w = tid >> 5;
    int m0 = blockIdx.y << 4;
    int n0 = (blockIdx.x << 7) + (w << 4);
    int nt = n0 >> 4;
    const int kth = H_ >> 5;

    stage_tile_32k(ldsU, u_t + (size_t)m0 * H_, tid);
    __syncthreads();

    v8f accz = {}, accr = {};
    v16bf bzc = load_b_frag(WzU, nt * kth, lane);
    v16bf brc = load_b_frag(WrU, nt * kth, lane);
    for (int kt = 0; kt < kth; kt++) {
        int ktn = (kt + 1 < kth) ? kt + 1 : kt;
        v16bf bzn = load_b_frag(WzU, nt * kth + ktn, lane);
        v16bf brn = load_b_frag(WrU, nt * kth + ktn, lane);
        v16bf a = load_a_lds(ldsU, kt << 5, lane);
        accz = wmma_bf16(a, bzc, accz);
        accr = wmma_bf16(a, brc, accr);
        bzc = bzn; brc = brn;
    }
    bzc = load_b_frag(WzH, nt * kth, lane);
    brc = load_b_frag(WrH, nt * kth, lane);
    for (int kt = 0; kt < kth; kt++) {
        int ktn = (kt + 1 < kth) ? kt + 1 : kt;
        v16bf bzn = load_b_frag(WzH, nt * kth + ktn, lane);
        v16bf brn = load_b_frag(WrH, nt * kth + ktn, lane);
        v16bf a = load_a_bf16(hBF, H_, m0, kt << 5, lane);
        accz = wmma_bf16(a, bzc, accz);
        accr = wmma_bf16(a, brc, accr);
        bzc = bzn; brc = brn;
    }
    int nl = lane & 15, gm = (lane >> 4) << 3;
    int n = n0 + nl;
    float bzv = bz[n], brv = br[n];
#pragma unroll
    for (int v = 0; v < 8; v++) {
        int m = m0 + gm + v;
        float zv = 1.0f / (1.0f + __expf(-(accz[v] + bzv)));
        float rv = 1.0f / (1.0f + __expf(-(accr[v] + brv)));
        zbuf[m * H_ + n] = zv;
        rh[m * H_ + n]   = f2bf(rv * hF32[m * H_ + n]);
    }
}

// ---- step 2: candidate + update; wave = 16(M) x 16(N) ----------------------
__global__ void step2_kernel(const unsigned short* __restrict__ u_t,
                             const unsigned short* __restrict__ rh,
                             const unsigned short* __restrict__ WiU,
                             const unsigned short* __restrict__ WiH,
                             const float* __restrict__ bi,
                             const float* __restrict__ zbuf,
                             float* __restrict__ hF32,
                             unsigned short* __restrict__ hBF) {
    __shared__ __align__(16) unsigned short ldsU[16 * H_];
    int tid = threadIdx.x, lane = tid & 31, w = tid >> 5;
    int m0 = blockIdx.y << 4;
    int n0 = (blockIdx.x << 7) + (w << 4);
    int nt = n0 >> 4;
    const int kth = H_ >> 5;

    stage_tile_32k(ldsU, u_t + (size_t)m0 * H_, tid);
    __syncthreads();

    v8f acc = {};
    v16bf bcur = load_b_frag(WiU, nt * kth, lane);
    for (int kt = 0; kt < kth; kt++) {
        int ktn = (kt + 1 < kth) ? kt + 1 : kt;
        v16bf bnxt = load_b_frag(WiU, nt * kth + ktn, lane);
        v16bf a = load_a_lds(ldsU, kt << 5, lane);
        acc = wmma_bf16(a, bcur, acc);
        bcur = bnxt;
    }
    bcur = load_b_frag(WiH, nt * kth, lane);
    for (int kt = 0; kt < kth; kt++) {
        int ktn = (kt + 1 < kth) ? kt + 1 : kt;
        v16bf bnxt = load_b_frag(WiH, nt * kth + ktn, lane);
        v16bf a = load_a_bf16(rh, H_, m0, kt << 5, lane);
        acc = wmma_bf16(a, bcur, acc);
        bcur = bnxt;
    }
    int nl = lane & 15, gm = (lane >> 4) << 3;
    int n = n0 + nl;
    float bin = bi[n];
#pragma unroll
    for (int v = 0; v < 8; v++) {
        int m = m0 + gm + v;
        float hp = tanhf(acc[v] + bin);
        float z  = zbuf[m * H_ + n];
        float ho = hF32[m * H_ + n];
        float hn = (1.0f - z) * ho + z * hp;
        hF32[m * H_ + n] = hn;
        hBF[m * H_ + n]  = f2bf(hn);
    }
}

extern "C" void kernel_launch(void* const* d_in, const int* in_sizes, int n_in,
                              void* d_out, int out_size, void* d_ws, size_t ws_size,
                              hipStream_t stream) {
    (void)in_sizes; (void)n_in; (void)out_size; (void)ws_size;
    const float* x  = (const float*)d_in[0];
    const float* Wm = (const float*)d_in[1];
    const float* bm = (const float*)d_in[2];
    const float* Wh = (const float*)d_in[3];
    const float* bh = (const float*)d_in[4];
    const float* Wz = (const float*)d_in[5];
    const float* bz = (const float*)d_in[6];
    const float* Wr = (const float*)d_in[7];
    const float* br = (const float*)d_in[8];
    const float* Wi = (const float*)d_in[9];
    const float* bi = (const float*)d_in[10];

    size_t off = 0;
    char* base = (char*)d_ws;
    auto alloc = [&](size_t bytes) -> char* {
        char* p = base + off;
        off += (bytes + 255) & ~(size_t)255;
        return p;
    };
    unsigned short* uBuf = (unsigned short*)alloc((size_t)S_ * B_ * H_ * 2);
    unsigned short* WmF  = (unsigned short*)alloc((size_t)H_ * D_ * 2);
    unsigned short* WhF  = (unsigned short*)alloc((size_t)H_ * D_ * 2);
    unsigned short* WzUf = (unsigned short*)alloc((size_t)H_ * H_ * 2);
    unsigned short* WzHf = (unsigned short*)alloc((size_t)H_ * H_ * 2);
    unsigned short* WrUf = (unsigned short*)alloc((size_t)H_ * H_ * 2);
    unsigned short* WrHf = (unsigned short*)alloc((size_t)H_ * H_ * 2);
    unsigned short* WiUf = (unsigned short*)alloc((size_t)H_ * H_ * 2);
    unsigned short* WiHf = (unsigned short*)alloc((size_t)H_ * H_ * 2);
    unsigned short* hBF  = (unsigned short*)alloc((size_t)B_ * H_ * 2);
    unsigned short* rhB  = (unsigned short*)alloc((size_t)B_ * H_ * 2);
    float*          zbuf = (float*)alloc((size_t)B_ * H_ * 4);
    float*          hF32 = (float*)d_out;

    wfrag_kernel<<<(H_ * D_) / 256, 256, 0, stream>>>(Wm, D_,      0,  H_, D_, WmF);
    wfrag_kernel<<<(H_ * D_) / 256, 256, 0, stream>>>(Wh, D_,      0,  H_, D_, WhF);
    wfrag_kernel<<<(H_ * H_) / 256, 256, 0, stream>>>(Wz, 2 * H_,  0,  H_, H_, WzUf);
    wfrag_kernel<<<(H_ * H_) / 256, 256, 0, stream>>>(Wz, 2 * H_,  H_, H_, H_, WzHf);
    wfrag_kernel<<<(H_ * H_) / 256, 256, 0, stream>>>(Wr, 2 * H_,  0,  H_, H_, WrUf);
    wfrag_kernel<<<(H_ * H_) / 256, 256, 0, stream>>>(Wr, 2 * H_,  H_, H_, H_, WrHf);
    wfrag_kernel<<<(H_ * H_) / 256, 256, 0, stream>>>(Wi, 2 * H_,  0,  H_, H_, WiUf);
    wfrag_kernel<<<(H_ * H_) / 256, 256, 0, stream>>>(Wi, 2 * H_,  H_, H_, H_, WiHf);

    h0_kernel<<<8, 256, 0, stream>>>(x, WhF, bh, hF32, hBF);
    ugemm_kernel<<<dim3(H_ / 256, (B_ * S_) / 32), 256, 0, stream>>>(x, WmF, bm, uBuf);

    for (int t = 0; t < S_; t++) {
        const unsigned short* ut = uBuf + (size_t)t * B_ * H_;
        step1_kernel<<<dim3(H_ / 128, B_ / 16), 256, 0, stream>>>(
            ut, hBF, hF32, WzUf, WzHf, WrUf, WrHf, bz, br, zbuf, rhB);
        step2_kernel<<<dim3(H_ / 128, B_ / 16), 256, 0, stream>>>(
            ut, rhB, WiUf, WiHf, bi, zbuf, hF32, hBF);
    }
}
